// Attention_2594160247065
// MI455X (gfx1250) — matrix operand order repaired and checked by reading, not compiled, not run on hardware
//
#include <hip/hip_runtime.h>
#include <hip/hip_bf16.h>
#include <math.h>

// ---------------------------------------------------------------------------
// CDNA5 / gfx1250 bf16 WMMA implementation of the perceiver-style attention.
// Wave32 only. All matmuls use v_wmma_f32_16x16x32_bf16.
// v2: software-pipelined GEMM cores (loads for K-step k+1 in flight during
//     WMMAs of step k), 32x64 wave tiles for the 1024x1024 GEMMs, prefetch
//     of next K/V j-block in the flash-attention loop.
// ---------------------------------------------------------------------------

typedef __bf16 bf16;
typedef __attribute__((ext_vector_type(16))) __bf16 v16bf;
typedef __attribute__((ext_vector_type(8)))  float  v8f;

#define WMMA_BF16(a, b, c) \
  __builtin_amdgcn_wmma_f32_16x16x32_bf16(false, (a), false, (b), (short)0, (c), false, false)

// Problem dims (fixed by the reference)
#define B_   2
#define N_   2048
#define D_   1024
#define H_   16
#define DH_  64
#define C_   256
#define CD_  512
#define J_   2305          // C + 1 + N
#define INNER_ (H_ * DH_)  // 1024
#define NEG_MAX_ (-3.402823466e38f)

// ---------------------------------------------------------------------------
// Fragment loaders (contiguous k order: lane half h holds K 16h..16h+15, wave32)
// ---------------------------------------------------------------------------

// A fragment: 16x32 bf16, row-major src with leading dim ld, at tile origin p.
// lanes 0-15: row m=lane, halfs 0-15 = K 0..15
// lanes 16-31: row m=lane-16, halfs 0-15 = K 16..31
__device__ inline v16bf load_a_frag(const bf16* p, int ld) {
  int lane = threadIdx.x & 31;
  int m  = lane & 15;
  int kb = (lane < 16) ? 0 : 16;
  const bf16* r = p + (size_t)m * ld + kb;
  v16bf a;
#pragma unroll
  for (int h = 0; h < 8; ++h) a[h] = r[h];
#pragma unroll
  for (int h = 0; h < 8; ++h) a[8 + h] = r[8 + h];
  return a;
}

// B fragment: 32x16 (KxN) where storage is N-major: B(k,n) = base[n*ld + k].
// lanes 0-15: col n=lane, halfs = K 0..15 ; lanes 16-31: col n=lane-16, K 16..31
__device__ inline v16bf load_b_nk(const bf16* base, int ld) {
  int lane = threadIdx.x & 31;
  int n  = lane & 15;
  int kb = (lane < 16) ? 0 : 16;
  const bf16* r = base + (size_t)n * ld + kb;
  v16bf b;
#pragma unroll
  for (int h = 0; h < 16; ++h) b[h] = r[h];
  return b;
}

// Guarded variant: rows n valid when n < nvalid, k valid when k < kvalid.
__device__ inline v16bf load_b_nk_g(const bf16* base, int ld, int nvalid, int kvalid) {
  int lane = threadIdx.x & 31;
  int n  = lane & 15;
  int kb = (lane < 16) ? 0 : 16;
  v16bf b;
#pragma unroll
  for (int h = 0; h < 16; ++h) {
    int k = kb + h;
    float val = 0.f;
    if (n < nvalid && k < kvalid) val = (float)base[(size_t)n * ld + k];
    b[h] = (bf16)val;
  }
  return b;
}

// Pipelined per-wave GEMM: C(16x64) += A(16xK) * B(Kx64), B stored N-major.
__device__ inline void wave_gemm_16x64_pipe(const bf16* A, int lda,
                                            const bf16* Bt, int ldb, int Kdim,
                                            v8f acc[4]) {
  v16bf a = load_a_frag(A, lda);
  v16bf b[4];
#pragma unroll
  for (int t = 0; t < 4; ++t) b[t] = load_b_nk(Bt + (size_t)(t * 16) * ldb, ldb);
  for (int k0 = 32; k0 < Kdim; k0 += 32) {
    v16bf na = load_a_frag(A + k0, lda);
    v16bf nb[4];
#pragma unroll
    for (int t = 0; t < 4; ++t) nb[t] = load_b_nk(Bt + (size_t)(t * 16) * ldb + k0, ldb);
#pragma unroll
    for (int t = 0; t < 4; ++t) acc[t] = WMMA_BF16(a, b[t], acc[t]);
    a = na;
#pragma unroll
    for (int t = 0; t < 4; ++t) b[t] = nb[t];
  }
#pragma unroll
  for (int t = 0; t < 4; ++t) acc[t] = WMMA_BF16(a, b[t], acc[t]);
}

// Pipelined per-wave GEMM: C(32x64) += A(32xK) * B(Kx64). B-frags reused x2.
__device__ inline void wave_gemm_32x64_pipe(const bf16* A, int lda,
                                            const bf16* Bt, int ldb, int Kdim,
                                            v8f acc[2][4]) {
  v16bf a0 = load_a_frag(A, lda);
  v16bf a1 = load_a_frag(A + (size_t)16 * lda, lda);
  v16bf b[4];
#pragma unroll
  for (int t = 0; t < 4; ++t) b[t] = load_b_nk(Bt + (size_t)(t * 16) * ldb, ldb);
  for (int k0 = 32; k0 < Kdim; k0 += 32) {
    v16bf na0 = load_a_frag(A + k0, lda);
    v16bf na1 = load_a_frag(A + (size_t)16 * lda + k0, lda);
    v16bf nb[4];
#pragma unroll
    for (int t = 0; t < 4; ++t) nb[t] = load_b_nk(Bt + (size_t)(t * 16) * ldb + k0, ldb);
#pragma unroll
    for (int t = 0; t < 4; ++t) {
      acc[0][t] = WMMA_BF16(a0, b[t], acc[0][t]);
      acc[1][t] = WMMA_BF16(a1, b[t], acc[1][t]);
    }
    a0 = na0; a1 = na1;
#pragma unroll
    for (int t = 0; t < 4; ++t) b[t] = nb[t];
  }
#pragma unroll
  for (int t = 0; t < 4; ++t) {
    acc[0][t] = WMMA_BF16(a0, b[t], acc[0][t]);
    acc[1][t] = WMMA_BF16(a1, b[t], acc[1][t]);
  }
}

// ---------------------------------------------------------------------------
// LayerNorm kernels
// ---------------------------------------------------------------------------

__global__ __launch_bounds__(256)
void ln_rows_bf16_kernel(const float* __restrict__ in, const float* __restrict__ gamma,
                         const float* __restrict__ beta, bf16* __restrict__ out, int L) {
  int row = blockIdx.x;
  const float* r = in + (size_t)row * L;
  float s = 0.f, sq = 0.f;
  for (int i = threadIdx.x; i < L; i += blockDim.x) { float v = r[i]; s += v; sq += v * v; }
  __shared__ float sh1[256], sh2[256];
  sh1[threadIdx.x] = s; sh2[threadIdx.x] = sq;
  __syncthreads();
  for (int st = 128; st > 0; st >>= 1) {
    if ((int)threadIdx.x < st) { sh1[threadIdx.x] += sh1[threadIdx.x + st]; sh2[threadIdx.x] += sh2[threadIdx.x + st]; }
    __syncthreads();
  }
  float mean = sh1[0] / (float)L;
  float var  = sh2[0] / (float)L - mean * mean;
  float rstd = rsqrtf(var + 1e-5f);
  for (int i = threadIdx.x; i < L; i += blockDim.x) {
    float g  = gamma ? gamma[i] : 1.f;
    float bt = beta  ? beta[i]  : 0.f;
    out[(size_t)row * L + i] = (bf16)((r[i] - mean) * rstd * g + bt);
  }
}

__global__ __launch_bounds__(256)
void ln_rows_f32_kernel(const float* __restrict__ in, const float* __restrict__ gamma,
                        float* __restrict__ out, int L) {
  int row = blockIdx.x;
  const float* r = in + (size_t)row * L;
  float s = 0.f, sq = 0.f;
  for (int i = threadIdx.x; i < L; i += blockDim.x) { float v = r[i]; s += v; sq += v * v; }
  __shared__ float sh1[256], sh2[256];
  sh1[threadIdx.x] = s; sh2[threadIdx.x] = sq;
  __syncthreads();
  for (int st = 128; st > 0; st >>= 1) {
    if ((int)threadIdx.x < st) { sh1[threadIdx.x] += sh1[threadIdx.x + st]; sh2[threadIdx.x] += sh2[threadIdx.x + st]; }
    __syncthreads();
  }
  float mean = sh1[0] / (float)L;
  float var  = sh2[0] / (float)L - mean * mean;
  float rstd = rsqrtf(var + 1e-5f);
  for (int i = threadIdx.x; i < L; i += blockDim.x)
    out[(size_t)row * L + i] = (r[i] - mean) * rstd * gamma[i];
}

// f32 (KxN row-major) -> bf16 transposed (NxK row-major)
__global__ __launch_bounds__(256)
void convT_kernel(const float* __restrict__ W, bf16* __restrict__ Wt, int K, int N) {
  size_t idx = (size_t)blockIdx.x * blockDim.x + threadIdx.x;
  size_t total = (size_t)K * N;
  if (idx < total) {
    size_t k = idx / N, n = idx % N;
    Wt[n * (size_t)K + k] = (bf16)W[idx];
  }
}

// ---------------------------------------------------------------------------
// Projection GEMMs
// ---------------------------------------------------------------------------

// q = xn @ Wq, scaled by Dh^-0.5, stored (B,H,N,Dh) bf16. 32x64 tiles, 2048 waves.
__global__ __launch_bounds__(256)
void gemm_q_kernel(const bf16* __restrict__ xn, const bf16* __restrict__ WqT,
                   bf16* __restrict__ qbuf) {
  int w  = (blockIdx.x << 3) | (threadIdx.x >> 5);
  int mt = w >> 4, ct = w & 15;
  int row0 = mt * 32, col0 = ct * 64;
  v8f acc[2][4] = {};
  wave_gemm_32x64_pipe(xn + (size_t)row0 * D_, D_, WqT + (size_t)col0 * D_, D_, D_, acc);
  int lane = threadIdx.x & 31, n = lane & 15, mb = (lane < 16) ? 0 : 8;
#pragma unroll
  for (int s = 0; s < 2; ++s)
#pragma unroll
    for (int t = 0; t < 4; ++t)
#pragma unroll
      for (int r = 0; r < 8; ++r) {
        int c = col0 + 16 * t + n;
        int h = c >> 6, d = c & 63;
        int rm = row0 + 16 * s + mb + r;
        int bb = rm >> 11, nn = rm & 2047;
        qbuf[(((size_t)bb * H_ + h) * N_ + nn) * DH_ + d] = (bf16)(acc[s][t][r] * 0.125f);
      }
}

// [k|v] = xn @ Wkv -> K (B,J,64) rows 257.. / V transposed (B,64,J).  512 waves.
__global__ __launch_bounds__(256)
void gemm_kv_kernel(const bf16* __restrict__ xn, const bf16* __restrict__ WkvT,
                    bf16* __restrict__ Kbuf, bf16* __restrict__ Vt) {
  int w  = (blockIdx.x << 3) | (threadIdx.x >> 5);
  int mt = w >> 1, ct = w & 1;
  int row0 = mt * 16, col0 = ct * 64;
  v8f acc[4] = {};
  wave_gemm_16x64_pipe(xn + (size_t)row0 * D_, D_, WkvT + (size_t)col0 * D_, D_, D_, acc);
  int lane = threadIdx.x & 31, n = lane & 15, mb = (lane < 16) ? 0 : 8;
#pragma unroll
  for (int t = 0; t < 4; ++t)
#pragma unroll
    for (int r = 0; r < 8; ++r) {
      int c  = col0 + 16 * t + n;
      int rm = row0 + mb + r;
      int bb = rm >> 11, nn = rm & 2047;
      float v = acc[t][r];
      if (c < DH_) Kbuf[((size_t)bb * J_ + (C_ + 1) + nn) * DH_ + c] = (bf16)v;
      else         Vt[((size_t)bb * DH_ + (c - DH_)) * J_ + (C_ + 1) + nn] = (bf16)v;
    }
}

// [ck|cv] = cn @ Wctx + bctx -> K rows 0..255 / V cols 0..255.  64 waves.
__global__ __launch_bounds__(256)
void gemm_ctx_kernel(const bf16* __restrict__ cn, const bf16* __restrict__ WctxT,
                     const float* __restrict__ bctx,
                     bf16* __restrict__ Kbuf, bf16* __restrict__ Vt) {
  int w  = (blockIdx.x << 3) | (threadIdx.x >> 5);
  int mt = w >> 1, ct = w & 1;
  int row0 = mt * 16, col0 = ct * 64;
  v8f acc[4] = {};
  wave_gemm_16x64_pipe(cn + (size_t)row0 * CD_, CD_, WctxT + (size_t)col0 * CD_, CD_, CD_, acc);
  int lane = threadIdx.x & 31, n = lane & 15, mb = (lane < 16) ? 0 : 8;
#pragma unroll
  for (int t = 0; t < 4; ++t)
#pragma unroll
    for (int r = 0; r < 8; ++r) {
      int c  = col0 + 16 * t + n;
      int rm = row0 + mb + r;
      int bb = rm >> 8, cc = rm & 255;
      float v = acc[t][r] + bctx[c];
      if (c < DH_) Kbuf[((size_t)bb * J_ + cc) * DH_ + c] = (bf16)v;
      else         Vt[((size_t)bb * DH_ + (c - DH_)) * J_ + cc] = (bf16)v;
    }
}

// null kv at j = C_
__global__ void null_fill_kernel(const float* __restrict__ null_kv,
                                 bf16* __restrict__ Kbuf, bf16* __restrict__ Vt) {
  int t = threadIdx.x;  // 0..127
  int bb = t >> 6, d = t & 63;
  Kbuf[((size_t)bb * J_ + C_) * DH_ + d] = (bf16)null_kv[d];
  Vt[((size_t)bb * DH_ + d) * J_ + C_]  = (bf16)null_kv[DH_ + d];
}

// ---------------------------------------------------------------------------
// Flash attention: one wave per (b, h, 16-query tile). 4096 waves, 8/block.
// ---------------------------------------------------------------------------
__global__ __launch_bounds__(256)
void attn_kernel(const bf16* __restrict__ qbuf, const bf16* __restrict__ Kbuf,
                 const bf16* __restrict__ Vt, const float* __restrict__ bias,
                 const unsigned char* __restrict__ mask, bf16* __restrict__ Obuf) {
  __shared__ bf16 plds[8][16][32];  // per-wave P tile (C-layout -> A-layout bridge)

  int w  = (blockIdx.x << 3) | (threadIdx.x >> 5);
  int it = w & 127;
  int hh = (w >> 7) & 15;
  int bb = w >> 11;
  int i0 = it * 16;
  int lane = threadIdx.x & 31;
  int n  = lane & 15;
  int mb = (lane < 16) ? 0 : 8;
  int wv = threadIdx.x >> 5;

  const bf16* qp = qbuf + (((size_t)bb * H_ + hh) * N_ + i0) * DH_;
  v16bf qa0 = load_a_frag(qp, DH_);
  v16bf qa1 = load_a_frag(qp + 32, DH_);

  const bf16* Kb = Kbuf + (size_t)bb * J_ * DH_;
  const bf16* Vb = Vt + (size_t)bb * DH_ * J_;
  const unsigned char* mk = mask + (size_t)bb * (C_ + N_);
  const float* bp = bias + ((size_t)hh * N_ + i0 + mb) * J_;

  float rmax[8], rsum[8];
#pragma unroll
  for (int r = 0; r < 8; ++r) { rmax[r] = NEG_MAX_; rsum[r] = 0.f; }
  v8f oacc[4] = {};

  const int NJB = (J_ + 31) / 32;  // 73
  for (int jb = 0; jb < NJB; ++jb) {
    int j0 = jb * 32;

    // Prefetch next j-block of K and V into cache (global_prefetch_b8).
    if (j0 + 32 < J_) {
      __builtin_prefetch(Kb + (size_t)(j0 + 32 + lane) * DH_, 0, 3);
      __builtin_prefetch(Vb + (size_t)(2 * lane) * J_ + j0 + 32, 0, 3);
      __builtin_prefetch(Vb + (size_t)(2 * lane + 1) * J_ + j0 + 32, 0, 3);
    }

    // S = q @ K^T for a 16x32 j-block (two 16x16 fragments)
    v8f s[2] = {};
#pragma unroll
    for (int jt = 0; jt < 2; ++jt) {
      int jj = j0 + jt * 16;
      int nv = J_ - jj; if (nv < 0) nv = 0;
      v16bf kb0 = load_b_nk_g(Kb + (size_t)jj * DH_,      DH_, nv, 32);
      v16bf kb1 = load_b_nk_g(Kb + (size_t)jj * DH_ + 32, DH_, nv, 32);
      s[jt] = WMMA_BF16(qa0, kb0, s[jt]);
      s[jt] = WMMA_BF16(qa1, kb1, s[jt]);
    }

    // bias + mask
    float sv[2][8];
#pragma unroll
    for (int jt = 0; jt < 2; ++jt) {
      int j = j0 + jt * 16 + n;
      bool keep = (j < J_) && ((j == 0) || (mk[j - 1] != 0));
#pragma unroll
      for (int r = 0; r < 8; ++r) {
        float v = NEG_MAX_;
        if (keep) v = s[jt][r] + bp[(size_t)r * J_ + j];
        sv[jt][r] = v;
      }
    }

    // online softmax: new row max (reduce across the 16 lanes of the row group)
    float nmax[8];
#pragma unroll
    for (int r = 0; r < 8; ++r) {
      float lm = fmaxf(sv[0][r], sv[1][r]);
#pragma unroll
      for (int xm = 1; xm < 16; xm <<= 1) lm = fmaxf(lm, __shfl_xor(lm, xm, 32));
      nmax[r] = fmaxf(rmax[r], lm);
    }
#pragma unroll
    for (int r = 0; r < 8; ++r) {
      float corr = __expf(rmax[r] - nmax[r]);  // ==1 when both still NEG_MAX
      rmax[r] = nmax[r];
      rsum[r] *= corr;
#pragma unroll
      for (int t = 0; t < 4; ++t) oacc[t][r] *= corr;
    }

    // P = exp(S - max), stash to LDS, accumulate row sums
    float psum[8];
#pragma unroll
    for (int jt = 0; jt < 2; ++jt) {
      int j = j0 + jt * 16 + n;
      bool keep = (j < J_) && ((j == 0) || (mk[j - 1] != 0));
#pragma unroll
      for (int r = 0; r < 8; ++r) {
        float p = keep ? __expf(sv[jt][r] - rmax[r]) : 0.f;
        plds[wv][mb + r][jt * 16 + n] = (bf16)p;
        if (jt == 0) psum[r] = p; else psum[r] += p;
      }
    }
#pragma unroll
    for (int r = 0; r < 8; ++r) {
      float ps = psum[r];
#pragma unroll
      for (int xm = 1; xm < 16; xm <<= 1) ps += __shfl_xor(ps, xm, 32);
      rsum[r] += ps;
    }

    __builtin_amdgcn_wave_barrier();
    v16bf pa = load_a_frag(&plds[wv][0][0], 32);
    __builtin_amdgcn_wave_barrier();

    // O += P @ V  (V transposed: (64, J))
    int kv = J_ - j0; if (kv > 32) kv = 32;
#pragma unroll
    for (int t = 0; t < 4; ++t) {
      v16bf vb = load_b_nk_g(Vb + (size_t)(t * 16) * J_ + j0, J_, 16, kv);
      oacc[t] = WMMA_BF16(pa, vb, oacc[t]);
    }
  }

  // store O[b, i, h*64 + d] = oacc / rsum
  bf16* op = Obuf + ((size_t)bb * N_ + i0 + mb) * INNER_ + (size_t)hh * DH_ + n;
#pragma unroll
  for (int t = 0; t < 4; ++t)
#pragma unroll
    for (int r = 0; r < 8; ++r)
      op[(size_t)r * INNER_ + 16 * t] = (bf16)(oacc[t][r] / rsum[r]);
}

// Y = O @ Wout (f32 out). 32x64 tiles, 2048 waves.
__global__ __launch_bounds__(256)
void gemm_out_kernel(const bf16* __restrict__ O, const bf16* __restrict__ WoutT,
                     float* __restrict__ Y) {
  int w  = (blockIdx.x << 3) | (threadIdx.x >> 5);
  int mt = w >> 4, ct = w & 15;
  int row0 = mt * 32, col0 = ct * 64;
  v8f acc[2][4] = {};
  wave_gemm_32x64_pipe(O + (size_t)row0 * INNER_, INNER_,
                       WoutT + (size_t)col0 * INNER_, INNER_, INNER_, acc);
  int lane = threadIdx.x & 31, n = lane & 15, mb = (lane < 16) ? 0 : 8;
#pragma unroll
  for (int s = 0; s < 2; ++s)
#pragma unroll
    for (int t = 0; t < 4; ++t)
#pragma unroll
      for (int r = 0; r < 8; ++r) {
        int c  = col0 + 16 * t + n;
        int rm = row0 + 16 * s + mb + r;
        Y[(size_t)rm * D_ + c] = acc[s][t][r];
      }
}

// ---------------------------------------------------------------------------
// Host launch
// ---------------------------------------------------------------------------
extern "C" void kernel_launch(void* const* d_in, const int* in_sizes, int n_in,
                              void* d_out, int out_size, void* d_ws, size_t ws_size,
                              hipStream_t stream) {
  (void)in_sizes; (void)n_in; (void)out_size; (void)ws_size;
  const float* x          = (const float*)d_in[0];
  const float* context    = (const float*)d_in[1];
  const unsigned char* mask = (const unsigned char*)d_in[2];  // jnp bool -> 1 byte
  const float* attn_bias  = (const float*)d_in[3];
  const float* norm_gamma = (const float*)d_in[4];
  const float* null_kv    = (const float*)d_in[5];
  const float* Wq         = (const float*)d_in[6];
  const float* Wkv        = (const float*)d_in[7];
  const float* ctx_ln_w   = (const float*)d_in[8];
  const float* ctx_ln_b   = (const float*)d_in[9];
  const float* Wctx       = (const float*)d_in[10];
  const float* bctx       = (const float*)d_in[11];
  const float* Wout       = (const float*)d_in[12];
  const float* out_gamma  = (const float*)d_in[13];
  float* out = (float*)d_out;

  // Workspace carve (~48 MB total)
  char* base = (char*)d_ws;
  size_t off = 0;
  auto take = [&](size_t bytes) -> void* {
    void* p = base + off;
    off += (bytes + 255) & ~(size_t)255;
    return p;
  };
  bf16*  xn    = (bf16*)take((size_t)B_ * N_ * D_ * sizeof(bf16));
  bf16*  cn    = (bf16*)take((size_t)B_ * C_ * CD_ * sizeof(bf16));
  bf16*  WqT   = (bf16*)take((size_t)D_ * INNER_ * sizeof(bf16));
  bf16*  WkvT  = (bf16*)take((size_t)D_ * 2 * DH_ * sizeof(bf16));
  bf16*  WctxT = (bf16*)take((size_t)CD_ * 2 * DH_ * sizeof(bf16));
  bf16*  WoutT = (bf16*)take((size_t)INNER_ * D_ * sizeof(bf16));
  bf16*  qbuf  = (bf16*)take((size_t)B_ * H_ * N_ * DH_ * sizeof(bf16));
  bf16*  Kbuf  = (bf16*)take((size_t)B_ * J_ * DH_ * sizeof(bf16));
  bf16*  Vt    = (bf16*)take((size_t)B_ * DH_ * J_ * sizeof(bf16));
  bf16*  Obuf  = (bf16*)take((size_t)B_ * N_ * INNER_ * sizeof(bf16));
  float* Ybuf  = (float*)take((size_t)B_ * N_ * D_ * sizeof(float));

  // 1. LayerNorms -> bf16
  ln_rows_bf16_kernel<<<B_ * N_, 256, 0, stream>>>(x, norm_gamma, nullptr, xn, D_);
  ln_rows_bf16_kernel<<<B_ * C_, 256, 0, stream>>>(context, ctx_ln_w, ctx_ln_b, cn, CD_);

  // 2. Weight conversion (bf16, transposed to N-major for contiguous B-fragments)
  convT_kernel<<<(D_ * INNER_ + 255) / 256, 256, 0, stream>>>(Wq, WqT, D_, INNER_);
  convT_kernel<<<(D_ * 2 * DH_ + 255) / 256, 256, 0, stream>>>(Wkv, WkvT, D_, 2 * DH_);
  convT_kernel<<<(CD_ * 2 * DH_ + 255) / 256, 256, 0, stream>>>(Wctx, WctxT, CD_, 2 * DH_);
  convT_kernel<<<(INNER_ * D_ + 255) / 256, 256, 0, stream>>>(Wout, WoutT, INNER_, D_);

  // 3. Projections (WMMA)
  gemm_q_kernel<<<256, 256, 0, stream>>>(xn, WqT, qbuf);      // 2048 waves, 32x64 tiles
  gemm_kv_kernel<<<64, 256, 0, stream>>>(xn, WkvT, Kbuf, Vt); // 512 waves
  gemm_ctx_kernel<<<8, 256, 0, stream>>>(cn, WctxT, bctx, Kbuf, Vt);
  null_fill_kernel<<<1, 128, 0, stream>>>(null_kv, Kbuf, Vt);

  // 4. Flash attention (WMMA, single pass over J)
  attn_kernel<<<(B_ * H_ * (N_ / 16)) / 8, 256, 0, stream>>>(qbuf, Kbuf, Vt,
                                                             attn_bias, mask, Obuf);

  // 5. Output projection + final LayerNorm
  gemm_out_kernel<<<256, 256, 0, stream>>>(Obuf, WoutT, Ybuf);
  ln_rows_f32_kernel<<<B_ * N_, 256, 0, stream>>>(Ybuf, out_gamma, out, D_);
}
